// _Attention_71176198029783
// MI455X (gfx1250) — matrix operand order repaired and checked by reading, not compiled, not run on hardware
//
#include <hip/hip_runtime.h>

// Problem constants
#define NPIX 2304            // 48*48
#define CCH  512
#define HID  512
#define NHEAD 8
#define DHEAD 64
#define MTOT 9216            // B * NPIX

typedef __attribute__((ext_vector_type(16))) _Float16 v16h;
typedef __attribute__((ext_vector_type(8)))  float    v8f;

union FragA { v16h v; uint4 u[2]; };
union Acc8  { v8f  v; float f[8]; };
union H16   { uint4 u[2]; _Float16 h[16]; };

__device__ __forceinline__ v8f wmma_f16(v16h a, v16h b, v8f c) {
  // 8 args: (neg_a, A, neg_b, B, c_mod, C, reuse_a, reuse_b)
  return __builtin_amdgcn_wmma_f32_16x16x32_f16(false, a, false, b, (short)0, c,
                                                false, false);
}

// Low 32 bits of a generic pointer into LDS == workgroup-relative LDS offset
__device__ __forceinline__ unsigned lds_addr(const void* p) {
  return (unsigned)(uintptr_t)p;
}

// Async global->LDS copy of 16 bytes; INST_OFFSET applies to BOTH addresses.
#define ASYNC_CP16(ldsoff, gptr, imm)                                         \
  asm volatile("global_load_async_to_lds_b128 %0, %1, off offset:" #imm      \
               :: "v"(ldsoff), "v"(gptr) : "memory")

__device__ __forceinline__ void wait_async0() {
  asm volatile("s_wait_asynccnt 0x0" ::: "memory");
}
__device__ __forceinline__ void wait_ds0() {
  asm volatile("s_wait_dscnt 0x0" ::: "memory");
}

// ---------------------------------------------------------------------------
// Kernel 1: per-pixel RMS norm over channels, output f16 [MTOT][512] row-major
// ---------------------------------------------------------------------------
__global__ __launch_bounds__(256)
void k_rms(const float* __restrict__ x, const float* __restrict__ g,
           _Float16* __restrict__ xn) {
  __shared__ float red[256];
  const int pix = blockIdx.x;                 // 0..9215
  const int b = pix / NPIX, p = pix % NPIX;
  const int t = threadIdx.x;
  const float* xb = x + (size_t)b * CCH * NPIX + p;
  float a0 = xb[(size_t)t * NPIX];
  float a1 = xb[(size_t)(t + 256) * NPIX];
  red[t] = a0 * a0 + a1 * a1;
  __syncthreads();
  for (int s = 128; s > 0; s >>= 1) {
    if (t < s) red[t] += red[t + s];
    __syncthreads();
  }
  float nrm = sqrtf(red[0]);
  float fac = 22.627416997969522f / fmaxf(nrm, 1e-12f);  // sqrt(512)/||x||
  _Float16* orow = xn + (size_t)pix * CCH;
  orow[t]       = (_Float16)(a0 * g[t] * fac);
  orow[t + 256] = (_Float16)(a1 * g[t + 256] * fac);
}

// ---------------------------------------------------------------------------
// Kernel 2: convert both weight matrices to f16 (done once per launch)
// ---------------------------------------------------------------------------
__global__ __launch_bounds__(256)
void k_wcvt(const float* __restrict__ wq, const float* __restrict__ wo,
            _Float16* __restrict__ wq16, _Float16* __restrict__ wo16) {
  const int i = blockIdx.x * 256 + threadIdx.x;
  const int nq = 3 * HID * CCH;               // 786432
  if (i < nq) wq16[i] = (_Float16)wq[i];
  else        wo16[i - nq] = (_Float16)wo[i - nq];
}

// ---------------------------------------------------------------------------
// Kernel 3: QKV GEMM  [9216 x 512] x [1536 x 512]^T, WMMA f16 -> f32.
// Double-buffered LDS, async global->LDS copies pipelined against WMMA.
// Epilogue scatters to q/k/v [b][head][N][64] f16; q pre-scaled by 1/8.
// ---------------------------------------------------------------------------
__global__ __launch_bounds__(128)
void k_qkv(const _Float16* __restrict__ A, const _Float16* __restrict__ W,
           _Float16* __restrict__ qb, _Float16* __restrict__ kb,
           _Float16* __restrict__ vb) {
  __shared__ __align__(16) _Float16 As[2][128][40];   // +8 pad per row
  __shared__ __align__(16) _Float16 Bs[2][64][40];
  const int t    = threadIdx.x;
  const int lane = t & 31, wave = t >> 5;
  const int hl   = lane >> 4, lrow = lane & 15;
  const int wm   = wave >> 1, wn = wave & 1;
  const int m0   = blockIdx.x * 128, n0 = blockIdx.y * 64;

  const v8f vzero = {0.f,0.f,0.f,0.f,0.f,0.f,0.f,0.f};
  Acc8 acc[4][2];
#pragma unroll
  for (int i = 0; i < 4; ++i)
#pragma unroll
    for (int j = 0; j < 2; ++j) acc[i][j].v = vzero;

  auto issueTile = [&](int k0, int buf) {
    // A tile 128x32: one row per thread (64B async, direct to LDS)
    unsigned la = lds_addr(&As[buf][t][0]);
    const _Float16* ga = A + (size_t)(m0 + t) * CCH + k0;
    ASYNC_CP16(la, ga, 0);  ASYNC_CP16(la, ga, 16);
    ASYNC_CP16(la, ga, 32); ASYNC_CP16(la, ga, 48);
    // B tile 64x32: two threads per row
    const int r = t >> 1, h = (t & 1) * 16;
    unsigned lb = lds_addr(&Bs[buf][r][h]);
    const _Float16* gb = W + (size_t)(n0 + r) * CCH + k0 + h;
    ASYNC_CP16(lb, gb, 0);  ASYNC_CP16(lb, gb, 16);
  };

  issueTile(0, 0);
  wait_async0();
  __syncthreads();

  for (int idx = 0; idx < 16; ++idx) {
    const int buf = idx & 1;
    if (idx < 15) issueTile((idx + 1) * 32, buf ^ 1);   // overlaps WMMA below

    FragA fb[2];
#pragma unroll
    for (int nj = 0; nj < 2; ++nj) {   // B frag: lane = column, K split by half
      const _Float16* bp = &Bs[buf][wn * 32 + nj * 16 + lrow][hl * 8];
      fb[nj].u[0] = *(const uint4*)bp;
      fb[nj].u[1] = *(const uint4*)(bp + 16);
    }
#pragma unroll
    for (int mi = 0; mi < 4; ++mi) {   // A frag: lane = row, K pairs in VGPRs
      FragA fa;
      const _Float16* ap = &As[buf][wm * 64 + mi * 16 + lrow][hl * 8];
      fa.u[0] = *(const uint4*)ap;
      fa.u[1] = *(const uint4*)(ap + 16);
#pragma unroll
      for (int nj = 0; nj < 2; ++nj)
        acc[mi][nj].v = wmma_f16(fa.v, fb[nj].v, acc[mi][nj].v);
    }
    if (idx < 15) { wait_async0(); __syncthreads(); }
  }

  // Epilogue: C/D layout -> scatter into q/k/v head layout
#pragma unroll
  for (int mi = 0; mi < 4; ++mi) {
    const int mbase = m0 + wm * 64 + mi * 16 + hl * 8;
#pragma unroll
    for (int nj = 0; nj < 2; ++nj) {
      const int o = n0 + wn * 32 + nj * 16 + lrow;
      const int which = o >> 9;            // 0=q 1=k 2=v
      const int hd = (o & 511) >> 6;
      const int d = o & 63;
      _Float16* dst = (which == 0) ? qb : ((which == 1) ? kb : vb);
      const float sc = (which == 0) ? 0.125f : 1.0f;   // fold 1/sqrt(64) into q
#pragma unroll
      for (int r = 0; r < 8; ++r) {
        const int mm = mbase + r;
        const int b = mm / NPIX, p = mm % NPIX;
        dst[(((size_t)b * NHEAD + hd) * NPIX + p) * DHEAD + d] =
            (_Float16)(acc[mi][nj].f[r] * sc);
      }
    }
  }
}

// ---------------------------------------------------------------------------
// Kernel 4: flash attention per (b, head, 64-query tile).
// 4 waves x 16 query rows; key tiles of 32, double-buffered:
// K via async copies, V loaded to regs early / transposed-stored after compute.
// ---------------------------------------------------------------------------
__global__ __launch_bounds__(128)
void k_attn(const _Float16* __restrict__ qb, const _Float16* __restrict__ kb,
            const _Float16* __restrict__ vb, _Float16* __restrict__ ao) {
  __shared__ __align__(16) _Float16 Qs[64][72];         // 64 q x 64 d (+8 pad)
  __shared__ __align__(16) _Float16 Ks[2][32][72];      // 32 k x 64 d
  __shared__ __align__(16) _Float16 Vt[2][64][40];      // transposed: [d][k]
  __shared__ __align__(16) _Float16 Ps[4][16][40];      // per-wave P (16q x 32k)

  const int t    = threadIdx.x;
  const int lane = t & 31, wave = t >> 5;
  const int hl   = lane >> 4, lrow = lane & 15;
  const int q0   = blockIdx.x * 64;
  const int hd   = blockIdx.y, b = blockIdx.z;
  const size_t base = ((size_t)b * NHEAD + hd) * NPIX * DHEAD;

  auto issueK = [&](int kt, int buf) {
    const int r = t >> 2, qd = (t & 3) * 16;
    unsigned lk = lds_addr(&Ks[buf][r][qd]);
    const _Float16* gk = kb + base + (size_t)(kt + r) * DHEAD + qd;
    ASYNC_CP16(lk, gk, 0); ASYNC_CP16(lk, gk, 16);
  };
  auto loadV = [&](int kt, H16& tmp) {
    const int r = t >> 2, qd = (t & 3) * 16;
    const uint4* src = (const uint4*)(vb + base + (size_t)(kt + r) * DHEAD + qd);
    tmp.u[0] = src[0]; tmp.u[1] = src[1];
  };
  auto storeV = [&](int buf, const H16& tmp) {
    const int r = t >> 2, qd = (t & 3) * 16;
#pragma unroll
    for (int j = 0; j < 16; ++j) Vt[buf][qd + j][r] = tmp.h[j];
  };

  // Prologue: Q tile + tile0 K async; tile0 V through regs
  {
    const int r = t >> 1, hh = (t & 1) * 32;
    unsigned lq = lds_addr(&Qs[r][hh]);
    const _Float16* gq = qb + base + (size_t)(q0 + r) * DHEAD + hh;
    ASYNC_CP16(lq, gq, 0);  ASYNC_CP16(lq, gq, 16);
    ASYNC_CP16(lq, gq, 32); ASYNC_CP16(lq, gq, 48);
  }
  issueK(0, 0);
  H16 v0tmp;
  loadV(0, v0tmp);
  storeV(0, v0tmp);
  wait_async0();
  __syncthreads();

  const v8f vzero = {0.f,0.f,0.f,0.f,0.f,0.f,0.f,0.f};
  Acc8 oacc[4];
#pragma unroll
  for (int i = 0; i < 4; ++i) oacc[i].v = vzero;
  float mrow[8], lsum[8];
#pragma unroll
  for (int r = 0; r < 8; ++r) { mrow[r] = -3.0e38f; lsum[r] = 0.f; }

  for (int idx = 0; idx < 72; ++idx) {
    const int buf = idx & 1;
    H16 vnext;
    if (idx < 71) {                       // prefetch next tile under compute
      issueK((idx + 1) * 32, buf ^ 1);
      loadV((idx + 1) * 32, vnext);
    }

    // S = Q K^T  (16 q rows x 32 keys), K-dim = 64 -> two k=32 WMMAs
    Acc8 s[2];
    s[0].v = vzero; s[1].v = vzero;
#pragma unroll
    for (int dk = 0; dk < DHEAD; dk += 32) {
      FragA fq;
      const _Float16* qp = &Qs[wave * 16 + lrow][dk + hl * 8];
      fq.u[0] = *(const uint4*)qp; fq.u[1] = *(const uint4*)(qp + 16);
#pragma unroll
      for (int nj = 0; nj < 2; ++nj) {
        FragA fk;
        const _Float16* kp = &Ks[buf][nj * 16 + lrow][dk + hl * 8];
        fk.u[0] = *(const uint4*)kp; fk.u[1] = *(const uint4*)(kp + 16);
        s[nj].v = wmma_f16(fq.v, fk.v, s[nj].v);
      }
    }

    // Online softmax: VGPR r of C/D layout holds row (r + hl*8); reduce the
    // key dimension across the 16 lanes of each half-wave.
#pragma unroll
    for (int r = 0; r < 8; ++r) {
      float v = fmaxf(s[0].f[r], s[1].f[r]);
#pragma unroll
      for (int m2 = 1; m2 < 16; m2 <<= 1) v = fmaxf(v, __shfl_xor(v, m2, 32));
      const float mn = fmaxf(mrow[r], v);
      const float alpha = __expf(mrow[r] - mn);
      const float p0 = __expf(s[0].f[r] - mn);
      const float p1 = __expf(s[1].f[r] - mn);
      float rs = p0 + p1;
#pragma unroll
      for (int m2 = 1; m2 < 16; m2 <<= 1) rs += __shfl_xor(rs, m2, 32);
      lsum[r] = lsum[r] * alpha + rs;
      mrow[r] = mn;
#pragma unroll
      for (int i = 0; i < 4; ++i) oacc[i].f[r] *= alpha;
      Ps[wave][r + hl * 8][lrow]      = (_Float16)p0;
      Ps[wave][r + hl * 8][16 + lrow] = (_Float16)p1;
    }
    wait_ds0();   // P written cross-lane within this wave; drain before reread

    // O += P @ V
    FragA fp;
    const _Float16* pp = &Ps[wave][lrow][hl * 8];
    fp.u[0] = *(const uint4*)pp; fp.u[1] = *(const uint4*)(pp + 16);
#pragma unroll
    for (int ni = 0; ni < 4; ++ni) {
      FragA fv;
      const _Float16* vp = &Vt[buf][ni * 16 + lrow][hl * 8];
      fv.u[0] = *(const uint4*)vp; fv.u[1] = *(const uint4*)(vp + 16);
      oacc[ni].v = wmma_f16(fp.v, fv.v, oacc[ni].v);
    }

    if (idx < 71) {
      storeV(buf ^ 1, vnext);   // buf^1 free since end of iter idx-1
      wait_async0();
      __syncthreads();
    }
  }

  // Normalize and store attn output as [b*NPIX + p][HID] f16 (col = head*64+d)
#pragma unroll
  for (int r = 0; r < 8; ++r) {
    const float inv = 1.0f / lsum[r];
    const int p = q0 + wave * 16 + hl * 8 + r;
    _Float16* dst = ao + ((size_t)(b * NPIX + p)) * HID + hd * DHEAD;
#pragma unroll
    for (int ni = 0; ni < 4; ++ni)
      dst[ni * 16 + lrow] = (_Float16)(oacc[ni].f[r] * inv);
  }
}

// ---------------------------------------------------------------------------
// Kernel 5: output projection [9216 x 512] x [512 x 512]^T + bias,
// async-pipelined like k_qkv; store float to d_out in [b][c][h][w] layout.
// ---------------------------------------------------------------------------
__global__ __launch_bounds__(128)
void k_out(const _Float16* __restrict__ A, const _Float16* __restrict__ W,
           const float* __restrict__ bias, float* __restrict__ y) {
  __shared__ __align__(16) _Float16 As[2][128][40];
  __shared__ __align__(16) _Float16 Bs[2][64][40];
  const int t    = threadIdx.x;
  const int lane = t & 31, wave = t >> 5;
  const int hl   = lane >> 4, lrow = lane & 15;
  const int wm   = wave >> 1, wn = wave & 1;
  const int m0   = blockIdx.x * 128, n0 = blockIdx.y * 64;

  const v8f vzero = {0.f,0.f,0.f,0.f,0.f,0.f,0.f,0.f};
  Acc8 acc[4][2];
#pragma unroll
  for (int i = 0; i < 4; ++i)
#pragma unroll
    for (int j = 0; j < 2; ++j) acc[i][j].v = vzero;

  auto issueTile = [&](int k0, int buf) {
    unsigned la = lds_addr(&As[buf][t][0]);
    const _Float16* ga = A + (size_t)(m0 + t) * HID + k0;
    ASYNC_CP16(la, ga, 0);  ASYNC_CP16(la, ga, 16);
    ASYNC_CP16(la, ga, 32); ASYNC_CP16(la, ga, 48);
    const int r = t >> 1, h = (t & 1) * 16;
    unsigned lb = lds_addr(&Bs[buf][r][h]);
    const _Float16* gb = W + (size_t)(n0 + r) * HID + k0 + h;
    ASYNC_CP16(lb, gb, 0);  ASYNC_CP16(lb, gb, 16);
  };

  issueTile(0, 0);
  wait_async0();
  __syncthreads();

  for (int idx = 0; idx < 16; ++idx) {
    const int buf = idx & 1;
    if (idx < 15) issueTile((idx + 1) * 32, buf ^ 1);

    FragA fb[2];
#pragma unroll
    for (int nj = 0; nj < 2; ++nj) {
      const _Float16* bp = &Bs[buf][wn * 32 + nj * 16 + lrow][hl * 8];
      fb[nj].u[0] = *(const uint4*)bp;
      fb[nj].u[1] = *(const uint4*)(bp + 16);
    }
#pragma unroll
    for (int mi = 0; mi < 4; ++mi) {
      FragA fa;
      const _Float16* ap = &As[buf][wm * 64 + mi * 16 + lrow][hl * 8];
      fa.u[0] = *(const uint4*)ap;
      fa.u[1] = *(const uint4*)(ap + 16);
#pragma unroll
      for (int nj = 0; nj < 2; ++nj)
        acc[mi][nj].v = wmma_f16(fa.v, fb[nj].v, acc[mi][nj].v);
    }
    if (idx < 15) { wait_async0(); __syncthreads(); }
  }

#pragma unroll
  for (int mi = 0; mi < 4; ++mi) {
    const int mbase = m0 + wm * 64 + mi * 16 + hl * 8;
#pragma unroll
    for (int nj = 0; nj < 2; ++nj) {
      const int o = n0 + wn * 32 + nj * 16 + lrow;
      const float bo = bias[o];
#pragma unroll
      for (int r = 0; r < 8; ++r) {
        const int mm = mbase + r;
        const int b = mm / NPIX, p = mm % NPIX;
        y[((size_t)b * CCH + o) * NPIX + p] = acc[mi][nj].f[r] + bo;
      }
    }
  }
}

// ---------------------------------------------------------------------------
extern "C" void kernel_launch(void* const* d_in, const int* in_sizes, int n_in,
                              void* d_out, int out_size, void* d_ws, size_t ws_size,
                              hipStream_t stream) {
  const float* x     = (const float*)d_in[0];
  const float* g     = (const float*)d_in[1];
  const float* w_qkv = (const float*)d_in[2];
  const float* w_out = (const float*)d_in[3];
  const float* b_out = (const float*)d_in[4];
  float* y = (float*)d_out;

  // workspace layout (bytes)
  char* ws = (char*)d_ws;
  _Float16* xn   = (_Float16*)(ws + 0);          // 9216*512 f16   = 9,437,184
  _Float16* qb   = (_Float16*)(ws + 9437184);    // 4*8*2304*64 f16
  _Float16* kb   = (_Float16*)(ws + 18874368);
  _Float16* vb   = (_Float16*)(ws + 28311552);
  _Float16* ao   = (_Float16*)(ws + 37748736);   // 9216*512 f16
  _Float16* wq16 = (_Float16*)(ws + 47185920);   // 1536*512 f16
  _Float16* wo16 = (_Float16*)(ws + 48758784);   // 512*512 f16

  k_rms <<<dim3(MTOT),        dim3(256), 0, stream>>>(x, g, xn);
  k_wcvt<<<dim3(4096),        dim3(256), 0, stream>>>(w_qkv, w_out, wq16, wo16);
  k_qkv <<<dim3(72, 24),      dim3(128), 0, stream>>>(xn, wq16, qb, kb, vb);
  k_attn<<<dim3(36, NHEAD, 4),dim3(128), 0, stream>>>(qb, kb, vb, ao);
  k_out <<<dim3(72, 8),       dim3(128), 0, stream>>>(ao, wo16, b_out, y);
}
